// GraphNet_1254130450915
// MI455X (gfx1250) — matrix operand order repaired and checked by reading, not compiled, not run on hardware
//
#include <hip/hip_runtime.h>

// ---------------------------------------------------------------------------
// GCN layer for MI455X (gfx1250, wave32, WMMA):
//   out = leaky_relu( X @ W_l  +  rownorm(adj) @ (X @ W_l1) ), slope 0.01
// B=32, N=D=768.  bf16 WMMA (v_wmma_f32_16x16x32_bf16) with f32 accumulate.
// Register double-buffered K-loop: global prefetch of tile k+1 overlaps the
// 8 WMMAs + LDS fragment loads of tile k.
// ---------------------------------------------------------------------------

typedef __attribute__((ext_vector_type(16))) __bf16 v16bf;
typedef __attribute__((ext_vector_type(8)))  __bf16 v8bf;
typedef __attribute__((ext_vector_type(8)))  float  v8f;
typedef __attribute__((ext_vector_type(4)))  float  v4f;

#define DD    768
#define NN    768
#define BB    32
#define TILE  128   // macro tile (M and N)
#define KSTEP 32    // WMMA K depth per step
#define LDA   40    // padded LDS stride (bf16 elems) for A rows (40*2B=80B)
#define LDB   40    // padded LDS stride (bf16 elems) for B cols

#define SHUF16(lo, hi) __builtin_shufflevector((lo), (hi), 0,1,2,3,4,5,6,7,8,9,10,11,12,13,14,15)

// ---------------------------------------------------------------------------
// Kernel 1: r_inv[row] = (rowsum == 0) ? 0 : 1/rowsum, one wave32 per row.
// ---------------------------------------------------------------------------
__global__ __launch_bounds__(256)
void rowsum_inv_kernel(const float* __restrict__ adj, float* __restrict__ r_inv) {
  const int wave = blockIdx.x * (blockDim.x >> 5) + (threadIdx.x >> 5);
  const int lane = threadIdx.x & 31;
  if (wave >= BB * NN) return;
  const float* row = adj + (size_t)wave * NN;
  float s = 0.f;
  for (int i = lane; i < NN; i += 32) s += row[i];
  #pragma unroll
  for (int off = 16; off > 0; off >>= 1) s += __shfl_xor(s, off, 32);
  if (lane == 0) r_inv[wave] = (s == 0.f) ? 0.f : 1.f / s;
}

// ---------------------------------------------------------------------------
// Fragment helpers (ISA 7.12.2 layouts).
//   A 16x32 bf16 : lane half h holds K {h*8..h*8+7, 16+h*8..16+h*8+7}, row = lane%16
//   B 32x16 bf16 : lane half h holds K {h*8..h*8+7, 16+h*8..16+h*8+7}, col = lane%16
//   C 16x16 f32  : lane holds col = lane%16, rows 8*(lane/16)+i for VGPR i
// LDS A tile: row-major  lA[row*LDA + k]   (rows 0..127, k 0..31)
// LDS B tile: col-major  lB[col*LDB + k]   (cols 0..127, k 0..31)
// ---------------------------------------------------------------------------
__device__ __forceinline__ v16bf frag_a(const __bf16* lA, int row, int lh) {
  const v8bf lo = *(const v8bf*)&lA[row * LDA + lh * 8];
  const v8bf hi = *(const v8bf*)&lA[row * LDA + 16 + lh * 8];
  return SHUF16(lo, hi);
}
__device__ __forceinline__ v16bf frag_b(const __bf16* lB, int col, int lh) {
  const v8bf lo = *(const v8bf*)&lB[col * LDB + lh * 8];
  const v8bf hi = *(const v8bf*)&lB[col * LDB + 16 + lh * 8];
  return SHUF16(lo, hi);
}

// ---------------------------------------------------------------------------
// Kernel 2: e1 = X @ W_l1   (X: [B*N, D] f32, W: [D, D] f32, out bf16)
// 128x128 tile per block, 8 waves as 4(m) x 2(n), each wave 32x64 = 2x4 WMMA.
// ---------------------------------------------------------------------------
__global__ __launch_bounds__(256)
void gemm_xw_bf16(const float* __restrict__ X, const float* __restrict__ W,
                  __bf16* __restrict__ Out) {
  __shared__ __align__(16) __bf16 lA[TILE * LDA];
  __shared__ __align__(16) __bf16 lB[TILE * LDB];

  const int m0 = blockIdx.y * TILE;
  const int n0 = blockIdx.x * TILE;
  const int t    = threadIdx.x;
  const int lane = t & 31, w = t >> 5;
  const int wm = w >> 1, wn = w & 1;          // 4 x 2 wave grid
  const int lh = lane >> 4, l16 = lane & 15;
  const int arow = t >> 1, akc = t & 1;       // A stage: 128 rows x 2 k-chunks
  const int bcol = t & 127, bkc = t >> 7;     // B stage: 128 cols x 2 k-chunks

  v8f acc[2][4] = {};
  v4f pa[4];      // 16 f32 of A (prefetch regs)
  float pb[16];   // 16 f32 of B (strided gather)

  auto gload = [&](int k0) {
    const v4f* ga = (const v4f*)(X + (size_t)(m0 + arow) * DD + k0 + akc * 16);
    #pragma unroll
    for (int u = 0; u < 4; u++) pa[u] = ga[u];
    const float* gb = W + (size_t)(k0 + bkc * 16) * DD + n0 + bcol;
    #pragma unroll
    for (int j = 0; j < 16; j++) pb[j] = gb[(size_t)j * DD];
  };
  auto lstore = [&]() {
    v8bf q0, q1;
    #pragma unroll
    for (int u = 0; u < 2; u++)
      #pragma unroll
      for (int v = 0; v < 4; v++) { q0[u * 4 + v] = (__bf16)pa[u][v];
                                    q1[u * 4 + v] = (__bf16)pa[2 + u][v]; }
    *(v8bf*)&lA[arow * LDA + akc * 16]     = q0;
    *(v8bf*)&lA[arow * LDA + akc * 16 + 8] = q1;
    #pragma unroll
    for (int j = 0; j < 8; j++) { q0[j] = (__bf16)pb[j]; q1[j] = (__bf16)pb[8 + j]; }
    *(v8bf*)&lB[bcol * LDB + bkc * 16]     = q0;
    *(v8bf*)&lB[bcol * LDB + bkc * 16 + 8] = q1;
  };

  gload(0);
  for (int k0 = 0; k0 < DD; k0 += KSTEP) {
    lstore();
    __syncthreads();
    if (k0 + KSTEP < DD) gload(k0 + KSTEP);   // overlaps compute below

    v16bf af[2], bfr[4];
    #pragma unroll
    for (int mt = 0; mt < 2; mt++) af[mt]  = frag_a(lA, wm * 32 + mt * 16 + l16, lh);
    #pragma unroll
    for (int nt = 0; nt < 4; nt++) bfr[nt] = frag_b(lB, wn * 64 + nt * 16 + l16, lh);
    #pragma unroll
    for (int mt = 0; mt < 2; mt++)
      #pragma unroll
      for (int nt = 0; nt < 4; nt++)
        acc[mt][nt] = __builtin_amdgcn_wmma_f32_16x16x32_bf16(
            false, af[mt], false, bfr[nt], (short)0, acc[mt][nt], false, false);
    __syncthreads();
  }

  // epilogue: bf16 store
  #pragma unroll
  for (int mt = 0; mt < 2; mt++)
    #pragma unroll
    for (int nt = 0; nt < 4; nt++) {
      const int c = n0 + wn * 64 + nt * 16 + l16;
      #pragma unroll
      for (int i = 0; i < 8; i++) {
        const int r = m0 + wm * 32 + mt * 16 + lh * 8 + i;
        Out[(size_t)r * DD + c] = (__bf16)acc[mt][nt][i];
      }
    }
}

// ---------------------------------------------------------------------------
// Kernel 3 (fused): out = leaky( X@W_l + (r_inv ⊙ adj) @ e1 )
// One accumulator set serves both GEMMs (row-normalization folded into the
// adj A-operand during LDS staging). Same register double buffering.
// ---------------------------------------------------------------------------
__global__ __launch_bounds__(256)
void gcn_fused(const float* __restrict__ X, const float* __restrict__ adj,
               const float* __restrict__ Wl, const __bf16* __restrict__ E1,
               const float* __restrict__ rinv, float* __restrict__ Out) {
  __shared__ __align__(16) __bf16 lA[TILE * LDA];
  __shared__ __align__(16) __bf16 lB[TILE * LDB];
  __shared__ float rv_s[TILE];

  const int b  = blockIdx.z;
  const int n0 = blockIdx.y * TILE;   // output row tile (within batch)
  const int d0 = blockIdx.x * TILE;   // output col tile
  const int t    = threadIdx.x;
  const int lane = t & 31, w = t >> 5;
  const int wm = w >> 1, wn = w & 1;
  const int lh = lane >> 4, l16 = lane & 15;
  const int arow = t >> 1, akc = t & 1;
  const int bcol = t & 127, bkc = t >> 7;

  if (t < TILE) rv_s[t] = rinv[b * NN + n0 + t];

  const float*  Xb = X   + (size_t)b * NN * DD;
  const float*  Ab = adj + (size_t)b * NN * NN;
  const __bf16* Eb = E1  + (size_t)b * NN * DD;

  v8f acc[2][4] = {};
  v4f pa[4];       // prefetch regs: A tile (f32)
  float pb[16];    // prefetch regs: B tile f32 (phase 1)
  __bf16 pe[16];   // prefetch regs: B tile bf16 (phase 2)

  auto store_a = [&](float scale) {
    v8bf q0, q1;
    #pragma unroll
    for (int u = 0; u < 2; u++)
      #pragma unroll
      for (int v = 0; v < 4; v++) { q0[u * 4 + v] = (__bf16)(pa[u][v] * scale);
                                    q1[u * 4 + v] = (__bf16)(pa[2 + u][v] * scale); }
    *(v8bf*)&lA[arow * LDA + akc * 16]     = q0;
    *(v8bf*)&lA[arow * LDA + akc * 16 + 8] = q1;
  };
  auto compute = [&]() {
    v16bf af[2], bfr[4];
    #pragma unroll
    for (int mt = 0; mt < 2; mt++) af[mt]  = frag_a(lA, wm * 32 + mt * 16 + l16, lh);
    #pragma unroll
    for (int nt = 0; nt < 4; nt++) bfr[nt] = frag_b(lB, wn * 64 + nt * 16 + l16, lh);
    #pragma unroll
    for (int mt = 0; mt < 2; mt++)
      #pragma unroll
      for (int nt = 0; nt < 4; nt++)
        acc[mt][nt] = __builtin_amdgcn_wmma_f32_16x16x32_bf16(
            false, af[mt], false, bfr[nt], (short)0, acc[mt][nt], false, false);
  };

  // ---- Phase 1: self transform  X[b] @ W_l ----
  auto gload1 = [&](int k0) {
    const v4f* ga = (const v4f*)(Xb + (size_t)(n0 + arow) * DD + k0 + akc * 16);
    #pragma unroll
    for (int u = 0; u < 4; u++) pa[u] = ga[u];
    const float* gb = Wl + (size_t)(k0 + bkc * 16) * DD + d0 + bcol;
    #pragma unroll
    for (int j = 0; j < 16; j++) pb[j] = gb[(size_t)j * DD];
  };
  gload1(0);
  for (int k0 = 0; k0 < DD; k0 += KSTEP) {
    store_a(1.0f);
    {
      v8bf q0, q1;
      #pragma unroll
      for (int j = 0; j < 8; j++) { q0[j] = (__bf16)pb[j]; q1[j] = (__bf16)pb[8 + j]; }
      *(v8bf*)&lB[bcol * LDB + bkc * 16]     = q0;
      *(v8bf*)&lB[bcol * LDB + bkc * 16 + 8] = q1;
    }
    __syncthreads();
    if (k0 + KSTEP < DD) gload1(k0 + KSTEP);
    compute();
    __syncthreads();
  }

  // ---- Phase 2: aggregate  (r_inv ⊙ adj[b]) @ e1[b] ----
  auto gload2 = [&](int m0) {
    const v4f* ga = (const v4f*)(Ab + (size_t)(n0 + arow) * NN + m0 + akc * 16);
    #pragma unroll
    for (int u = 0; u < 4; u++) pa[u] = ga[u];
    const __bf16* gb = Eb + (size_t)(m0 + bkc * 16) * DD + d0 + bcol;
    #pragma unroll
    for (int j = 0; j < 16; j++) pe[j] = gb[(size_t)j * DD];
  };
  const float rv = rv_s[arow];    // rv_s written pre-phase-1; barriers since
  gload2(0);
  for (int m0 = 0; m0 < NN; m0 += KSTEP) {
    store_a(rv);
    {
      v8bf q0, q1;
      #pragma unroll
      for (int j = 0; j < 8; j++) { q0[j] = pe[j]; q1[j] = pe[8 + j]; }
      *(v8bf*)&lB[bcol * LDB + bkc * 16]     = q0;
      *(v8bf*)&lB[bcol * LDB + bkc * 16 + 8] = q1;
    }
    __syncthreads();
    if (m0 + KSTEP < NN) gload2(m0 + KSTEP);
    compute();
    __syncthreads();
  }

  // ---- epilogue: leaky ReLU, f32 store ----
  #pragma unroll
  for (int mt = 0; mt < 2; mt++)
    #pragma unroll
    for (int nt = 0; nt < 4; nt++) {
      const int c = d0 + wn * 64 + nt * 16 + l16;
      #pragma unroll
      for (int i = 0; i < 8; i++) {
        const int r = n0 + wm * 32 + mt * 16 + lh * 8 + i;
        float v = acc[mt][nt][i];
        v = (v > 0.f) ? v : 0.01f * v;
        Out[((size_t)b * NN + r) * DD + c] = v;
      }
    }
}

// ---------------------------------------------------------------------------
extern "C" void kernel_launch(void* const* d_in, const int* in_sizes, int n_in,
                              void* d_out, int out_size, void* d_ws, size_t ws_size,
                              hipStream_t stream) {
  (void)in_sizes; (void)n_in; (void)out_size; (void)ws_size;
  const float* X    = (const float*)d_in[0];   // [32, 768, 768]
  const float* adj  = (const float*)d_in[1];   // [32, 768, 768]
  const float* W_l  = (const float*)d_in[2];   // [768, 768]
  const float* W_l1 = (const float*)d_in[3];   // [768, 768]
  float* out = (float*)d_out;

  // workspace: r_inv (B*N f32, padded to 128KB) then e1 (B*N*D bf16)
  float*  r_inv = (float*)d_ws;
  __bf16* e1    = (__bf16*)((char*)d_ws + (128u << 10));

  // 1) row-sum inverse: 32*768 rows, 8 waves/block
  rowsum_inv_kernel<<<dim3((BB * NN) / 8), 256, 0, stream>>>(adj, r_inv);

  // 2) e1 = X @ W_l1  (rows flattened: 24576 x 768)
  gemm_xw_bf16<<<dim3(DD / TILE, (BB * NN) / TILE), 256, 0, stream>>>(X, W_l1, e1);

  // 3) fused: out = leaky(X@W_l + rownorm(adj)@e1)
  gcn_fused<<<dim3(DD / TILE, NN / TILE, BB), 256, 0, stream>>>(X, adj, W_l, e1, r_inv, out);
}
